// MinkUNetBaseSpconv_83373905150633
// MI455X (gfx1250) — compile-verified, matrix operand order checked by reading, not run on hardware
//
#include <hip/hip_runtime.h>

#define BN_EPS 1e-5f
#define ZPFX 512   // zeroed bf16 prefix in front of every activation buffer

typedef __attribute__((ext_vector_type(16))) __bf16 v16bf;
typedef __attribute__((ext_vector_type(8)))  float  v8f;

union frag16 { v16bf v; uint4 q[2]; };   // 32 bytes: one WMMA 16-bit fragment

// ---------------- bf16 helpers (RNE float->bf16, bit moves) ----------------
__device__ __forceinline__ unsigned short f2bf(float f) {
  unsigned int u = __builtin_bit_cast(unsigned int, f);
  u += 0x7fffu + ((u >> 16) & 1u);          // round to nearest even
  return (unsigned short)(u >> 16);
}
__device__ __forceinline__ float bf2f(unsigned short s) {
  unsigned int u = ((unsigned int)s) << 16;
  return __builtin_bit_cast(float, u);
}

// ---------------------------------------------------------------------------
// Weight prepack: fp32 [Cout][Cin][k3]  ->  bf16 [k3][Cout][Cpad]
// ---------------------------------------------------------------------------
__global__ __launch_bounds__(256) void prepack_w_kernel(
    const float* __restrict__ w, unsigned short* __restrict__ wpk,
    int Cin, int Cpad, int Cout, int k3)
{
  long e = (long)blockIdx.x * 256 + threadIdx.x;
  long total = (long)k3 * Cout * Cpad;
  if (e >= total) return;
  int ci = (int)(e % Cpad);
  long r = e / Cpad;
  int n = (int)(r % Cout);
  int t = (int)(r / Cout);
  float v = (ci < Cin) ? w[((long)n * Cin + ci) * k3 + t] : 0.f;
  wpk[e] = f2bf(v);
}

// ------------------------- zero a small u16 region -------------------------
__global__ __launch_bounds__(256) void zero_u16_kernel(unsigned short* __restrict__ p, int n)
{
  int i = blockIdx.x * 256 + threadIdx.x;
  if (i < n) p[i] = 0;
}

// ---------------------------------------------------------------------------
// Implicit-GEMM 3D convolution, WMMA bf16, fp32 accumulate.
//   activations: bf16 channel-last [D*H*W][Cin], with ZPFX zeroed elements
//                in front (OOB taps read zeros via negative offset)
//   weights:     bf16 prepacked   [k3][Cout][Cin]
//   output:      fp32 channel-last [Mo][Cout]
// One wave = 32-voxel x 32-outchannel tile. Every fragment = two aligned
// global_load_b128 (K = channel = contiguous dim); no data/pointer selects
// in the inner loop — only a 32-bit offset select per tap.
// Fragment K-striping per CDNA5 ISA 7.12.2: lane half lh holds K elements
// {cb+8*lh .. +7} and {cb+16+8*lh .. +7}.
// ---------------------------------------------------------------------------
__global__ __launch_bounds__(32) void conv3d_bf16_wmma(
    const unsigned short* __restrict__ in, const unsigned short* __restrict__ wpk,
    float* __restrict__ out, int Cin, int Cout, int D, int H, int W,
    int Do, int Ho, int Wo, int k, int stride, int pad)
{
  const int lane = threadIdx.x & 31;
  const int lh = lane >> 4;
  const int ln = lane & 15;
  const int Mo = Do * Ho * Wo;
  const int mbase = blockIdx.x * 32;
  const int nbase = blockIdx.y * 32;
  const int k2 = k * k, k3 = k2 * k;

  // two A-row voxels per lane (rows replicated across lane halves)
  int mv0 = mbase + ln, mv1 = mbase + 16 + ln;
  bool ok0 = mv0 < Mo, ok1 = mv1 < Mo;
  int od0 = 0, oh0 = 0, ow0 = 0, od1 = 0, oh1 = 0, ow1 = 0;
  if (ok0) { od0 = mv0 / (Ho * Wo); int r = mv0 - od0 * Ho * Wo; oh0 = r / Wo; ow0 = r - oh0 * Wo; }
  if (ok1) { od1 = mv1 / (Ho * Wo); int r = mv1 - od1 * Ho * Wo; oh1 = r / Wo; ow1 = r - oh1 * Wo; }

  const int n0 = nbase + ln;
  const size_t wrow = (size_t)Cout * Cin;
  v8f c00 = {}, c01 = {}, c10 = {}, c11 = {};

  for (int t = 0; t < k3; ++t) {
    int kd = t / k2, rr = t - kd * k2, kh = rr / k, kw = rr - kh * k;
    int id0 = od0 * stride - pad + kd, ih0 = oh0 * stride - pad + kh, iw0 = ow0 * stride - pad + kw;
    int id1 = od1 * stride - pad + kd, ih1 = oh1 * stride - pad + kh, iw1 = ow1 * stride - pad + kw;
    bool in0 = ok0 && (unsigned)id0 < (unsigned)D && (unsigned)ih0 < (unsigned)H &&
               (unsigned)iw0 < (unsigned)W;
    bool in1 = ok1 && (unsigned)id1 < (unsigned)D && (unsigned)ih1 < (unsigned)H &&
               (unsigned)iw1 < (unsigned)W;
    // 32-bit offset select: OOB lanes read the zeroed prefix
    int aoff0 = in0 ? ((id0 * H + ih0) * W + iw0) * Cin : -ZPFX;
    int aoff1 = in1 ? ((id1 * H + ih1) * W + iw1) * Cin : -ZPFX;
    const unsigned short* bp0 = wpk + (size_t)t * wrow + (size_t)n0 * Cin;
    const unsigned short* bp1 = bp0 + (size_t)16 * Cin;

    for (int cb = 0; cb < Cin; cb += 32) {
      const int o0 = cb + lh * 8;       // first 8 contiguous K elems
      const int o1 = cb + 16 + lh * 8;  // second 8 contiguous K elems
      frag16 a0, a1, b0, b1;
      a0.q[0] = *(const uint4*)(in + aoff0 + o0);
      a0.q[1] = *(const uint4*)(in + aoff0 + o1);
      a1.q[0] = *(const uint4*)(in + aoff1 + o0);
      a1.q[1] = *(const uint4*)(in + aoff1 + o1);
      b0.q[0] = *(const uint4*)(bp0 + o0);
      b0.q[1] = *(const uint4*)(bp0 + o1);
      b1.q[0] = *(const uint4*)(bp1 + o0);
      b1.q[1] = *(const uint4*)(bp1 + o1);
      c00 = __builtin_amdgcn_wmma_f32_16x16x32_bf16(false, a0.v, false, b0.v, (short)0, c00, false, false);
      c01 = __builtin_amdgcn_wmma_f32_16x16x32_bf16(false, a0.v, false, b1.v, (short)0, c01, false, false);
      c10 = __builtin_amdgcn_wmma_f32_16x16x32_bf16(false, a1.v, false, b0.v, (short)0, c10, false, false);
      c11 = __builtin_amdgcn_wmma_f32_16x16x32_bf16(false, a1.v, false, b1.v, (short)0, c11, false, false);
    }
  }
#pragma unroll
  for (int j = 0; j < 8; ++j) {
    int m0 = mbase + lh * 8 + j;       // D layout: row = vgpr + 8*laneHalf
    int m1 = m0 + 16;
    if (m0 < Mo) {
      out[(size_t)m0 * Cout + n0] = c00[j];
      out[(size_t)m0 * Cout + n0 + 16] = c01[j];
    }
    if (m1 < Mo) {
      out[(size_t)m1 * Cout + n0] = c10[j];
      out[(size_t)m1 * Cout + n0 + 16] = c11[j];
    }
  }
}

// ---------------------------------------------------------------------------
// Transposed conv k=2 s=2: out[2i+p] = sum_ci x[i] * w[:, :, 1-p]  =>
// 8 parity-class 1x1 GEMMs (blockIdx.z = parity), K = Cin.
// ---------------------------------------------------------------------------
__global__ __launch_bounds__(32) void deconv2x_bf16_wmma(
    const unsigned short* __restrict__ in, const unsigned short* __restrict__ wpk,
    float* __restrict__ out, int Cin, int Cout, int Di, int Hi, int Wi)
{
  const int lane = threadIdx.x & 31;
  const int lh = lane >> 4;
  const int ln = lane & 15;
  const int Mi = Di * Hi * Wi;
  const int Ho = 2 * Hi, Wo = 2 * Wi;
  const int mbase = blockIdx.x * 32;
  const int nbase = blockIdx.y * 32;
  const int z = blockIdx.z;
  const int pd = (z >> 2) & 1, ph = (z >> 1) & 1, pw = z & 1;
  const int tap = ((1 - pd) * 2 + (1 - ph)) * 2 + (1 - pw);

  int mv0 = mbase + ln, mv1 = mbase + 16 + ln;
  bool ok0 = mv0 < Mi, ok1 = mv1 < Mi;
  int aoff0 = ok0 ? mv0 * Cin : -ZPFX;
  int aoff1 = ok1 ? mv1 * Cin : -ZPFX;
  const int n0 = nbase + ln;
  const unsigned short* bp0 = wpk + ((size_t)tap * Cout + n0) * Cin;
  const unsigned short* bp1 = bp0 + (size_t)16 * Cin;
  v8f c00 = {}, c01 = {}, c10 = {}, c11 = {};

  for (int cb = 0; cb < Cin; cb += 32) {
    const int o0 = cb + lh * 8;
    const int o1 = cb + 16 + lh * 8;
    frag16 a0, a1, b0, b1;
    a0.q[0] = *(const uint4*)(in + aoff0 + o0);
    a0.q[1] = *(const uint4*)(in + aoff0 + o1);
    a1.q[0] = *(const uint4*)(in + aoff1 + o0);
    a1.q[1] = *(const uint4*)(in + aoff1 + o1);
    b0.q[0] = *(const uint4*)(bp0 + o0);
    b0.q[1] = *(const uint4*)(bp0 + o1);
    b1.q[0] = *(const uint4*)(bp1 + o0);
    b1.q[1] = *(const uint4*)(bp1 + o1);
    c00 = __builtin_amdgcn_wmma_f32_16x16x32_bf16(false, a0.v, false, b0.v, (short)0, c00, false, false);
    c01 = __builtin_amdgcn_wmma_f32_16x16x32_bf16(false, a0.v, false, b1.v, (short)0, c01, false, false);
    c10 = __builtin_amdgcn_wmma_f32_16x16x32_bf16(false, a1.v, false, b0.v, (short)0, c10, false, false);
    c11 = __builtin_amdgcn_wmma_f32_16x16x32_bf16(false, a1.v, false, b1.v, (short)0, c11, false, false);
  }
#pragma unroll
  for (int j = 0; j < 8; ++j) {
    int m0 = mbase + lh * 8 + j;
    int m1 = m0 + 16;
    if (m0 < Mi) {
      int id = m0 / (Hi * Wi); int r = m0 - id * Hi * Wi; int ih = r / Wi; int iw = r - ih * Wi;
      size_t o = (size_t)((2 * id + pd) * Ho + (2 * ih + ph)) * Wo + (2 * iw + pw);
      out[o * Cout + n0] = c00[j];
      out[o * Cout + n0 + 16] = c01[j];
    }
    if (m1 < Mi) {
      int id = m1 / (Hi * Wi); int r = m1 - id * Hi * Wi; int ih = r / Wi; int iw = r - ih * Wi;
      size_t o = (size_t)((2 * id + pd) * Ho + (2 * ih + ph)) * Wo + (2 * iw + pw);
      out[o * Cout + n0] = c10[j];
      out[o * Cout + n0 + 16] = c11[j];
    }
  }
}

// ------------- BatchNorm stats: per-channel mean / biased var --------------
__global__ __launch_bounds__(256) void bn_stats_kernel(const float* __restrict__ x,
                                                       float* __restrict__ st, int C, int M)
{
  __shared__ float ss[256], sq[256];
  int c = blockIdx.x;
  float s = 0.f, q = 0.f;
  for (int i = threadIdx.x; i < M; i += 256) {
    float v = x[(long)i * C + c];
    s += v; q += v * v;
  }
  ss[threadIdx.x] = s; sq[threadIdx.x] = q;
  __syncthreads();
  for (int o = 128; o > 0; o >>= 1) {
    if (threadIdx.x < (unsigned)o) { ss[threadIdx.x] += ss[threadIdx.x + o];
                                     sq[threadIdx.x] += sq[threadIdx.x + o]; }
    __syncthreads();
  }
  if (threadIdx.x == 0) {
    float mean = ss[0] / (float)M;
    float var  = sq[0] / (float)M - mean * mean;
    st[c] = mean; st[C + c] = var;
  }
}

// --- fused BN (+ optional residual add, ReLU); bf16 channel-last out with
//     stride/offset (for concat placement) and/or fp32 channel-major out ----
__global__ __launch_bounds__(256) void bn_apply_kernel(
    const float* __restrict__ x, const float* __restrict__ st,
    const float* __restrict__ g, const float* __restrict__ b,
    const unsigned short* __restrict__ idn, unsigned short* __restrict__ obf,
    float* __restrict__ of, int C, int M, int Cstr, int coff, int relu)
{
  long i = (long)blockIdx.x * 256 + threadIdx.x;
  if (i >= (long)C * M) return;
  int c = (int)(i % C);
  long m = i / C;
  float y = g[c] * (x[i] - st[c]) * rsqrtf(st[C + c] + BN_EPS) + b[c];
  if (idn) y += bf2f(idn[i]);
  if (relu) y = fmaxf(y, 0.f);
  if (obf) obf[m * Cstr + coff + c] = f2bf(y);
  if (of)  of[(long)c * M + m] = y;       // channel-major fp32 (reference NCDHW)
}

// -------- skip-copy into concat buffer (channel-last interleave) -----------
__global__ __launch_bounds__(256) void concat_copy_kernel(
    const unsigned short* __restrict__ skip, unsigned short* __restrict__ cat,
    int Cskip, int Ctot, int coff, int M)
{
  long i = (long)blockIdx.x * 256 + threadIdx.x;
  if (i >= (long)M * Cskip) return;
  int c = (int)(i % Cskip);
  long m = i / Cskip;
  cat[m * Ctot + coff + c] = skip[i];
}

// --- input convert: fp32 [4][M] channel-major -> bf16 [M][32] zero-padded --
__global__ __launch_bounds__(256) void cvt_x_kernel(const float* __restrict__ x,
                                                    unsigned short* __restrict__ xb, int M)
{
  long e = (long)blockIdx.x * 256 + threadIdx.x;
  if (e >= (long)M * 32) return;
  int c = (int)(e % 32);
  long m = e / 32;
  xb[e] = (c < 4) ? f2bf(x[(long)c * M + m]) : (unsigned short)0;
}

// ------------- final 1x1x1 conv (96 -> 20, fp32 channel-major) -------------
__global__ __launch_bounds__(256) void final_conv_kernel(
    const float* __restrict__ feat, const float* __restrict__ w,
    const float* __restrict__ bias, float* __restrict__ out, int M)
{
  long i = (long)blockIdx.x * 256 + threadIdx.x;
  if (i >= (long)20 * M) return;
  int c = (int)(i / M);
  int m = (int)(i - (long)c * M);
  float s = bias[c];
#pragma unroll 4
  for (int ci = 0; ci < 96; ++ci) s += feat[(long)ci * M + m] * w[c * 96 + ci];
  out[i] = s;
}

// ===========================================================================
extern "C" void kernel_launch(void* const* d_in, const int* in_sizes, int n_in,
                              void* d_out, int out_size, void* d_ws, size_t ws_size,
                              hipStream_t stream)
{
  (void)in_sizes; (void)n_in; (void)out_size; (void)ws_size;

  auto Pf = [&](int i) { return (const float*)d_in[i]; };

  // ---- parameter indices (setup_inputs insertion order; x is d_in[0]) ----
  const int I_C0 = 1;
  const int I_CV[5] = {0, 4, 7, 10, 13};             // conv1..conv4
  const int I_TR[8] = {0, 0, 0, 0, 16, 19, 22, 25};  // tr4..tr7
  const int SBASE[8] = {28, 40, 55, 70, 85, 100, 115, 130};
  const bool SDOWN[8] = {false, true, true, true, true, true, true, true};
  const int SCIN[8] = {32, 32, 64, 128, 384, 192, 128, 128};
  const int SPL[8]  = {32, 64, 128, 256, 256, 128, 96, 96};
  const int I_FW = 145, I_FB = 146;

  const int M48 = 48 * 48 * 48, M24 = 24 * 24 * 24, M12 = 12 * 12 * 12, M6 = 216;

  // ---- workspace bump allocator ----
  char* wsb = (char*)d_ws;
  size_t off = 0;
  auto alloc = [&](size_t bytes) -> void* {
    void* p = wsb + off;
    off += (bytes + 255) & ~(size_t)255;
    return p;
  };
  // activation buffer: ZPFX zeroed bf16 elements in front for OOB reads
  auto allocAct = [&](size_t elems) -> unsigned short* {
    unsigned short* raw = (unsigned short*)alloc((elems + ZPFX) * 2);
    zero_u16_kernel<<<dim3((ZPFX + 255) / 256, 1, 1), dim3(256, 1, 1), 0, stream>>>(raw, ZPFX);
    return raw + ZPFX;
  };

  const size_t MAXE = (size_t)96 * M48;                     // largest layer output
  float* scrA = (float*)alloc(MAXE * 4);                    // main-path fp32 out
  float* scrB = (float*)alloc(MAXE * 4);                    // identity-path fp32 out
  unsigned short* wpk = (unsigned short*)alloc((size_t)27 * 256 * 384 * 2);  // prepack
  float* st1 = (float*)alloc(512 * 4);
  float* st2 = (float*)alloc(512 * 4);
  float* st3 = (float*)alloc(512 * 4);
  unsigned short* bfT = allocAct(MAXE);                 // block conv1 act
  unsigned short* bfI = allocAct(MAXE);                 // block identity act
  unsigned short* bfU = allocAct(MAXE);                 // block0 output
  unsigned short* xb = allocAct((size_t)32 * M48);      // padded input
  unsigned short* p1 = allocAct((size_t)32 * M48);      // skip (stem)
  unsigned short* b1 = allocAct((size_t)32 * M24);      // skip
  unsigned short* b2 = allocAct((size_t)64 * M12);      // skip
  unsigned short* b3 = allocAct((size_t)128 * M6);      // skip
  const size_t CATE = (size_t)128 * M48;
  unsigned short* bufX = allocAct(CATE);
  unsigned short* bufY = allocAct(CATE);

  // ---- launch helpers ----
  auto prepack = [&](int wi, int CinReal, int Cpad, int Cout, int k3) {
    long total = (long)k3 * Cout * Cpad;
    prepack_w_kernel<<<dim3((unsigned)((total + 255) / 256), 1, 1), dim3(256, 1, 1), 0,
                       stream>>>(Pf(wi), wpk, CinReal, Cpad, Cout, k3);
  };
  auto conv = [&](const unsigned short* in, int wi, float* out, int CinReal, int Cpad,
                  int Cout, int D, int k, int s, int p) {
    prepack(wi, CinReal, Cpad, Cout, k * k * k);
    int Do = (D + 2 * p - k) / s + 1;
    int Mo = Do * Do * Do;
    dim3 g((Mo + 31) / 32, Cout / 32, 1);
    conv3d_bf16_wmma<<<g, dim3(32, 1, 1), 0, stream>>>(in, wpk, out, Cpad, Cout, D, D, D,
                                                       Do, Do, Do, k, s, p);
  };
  auto deconv = [&](const unsigned short* in, int wi, float* out, int Cin, int Cout,
                    int Di) {
    prepack(wi, Cin, Cin, Cout, 8);
    int Mi = Di * Di * Di;
    dim3 g((Mi + 31) / 32, Cout / 32, 8);
    deconv2x_bf16_wmma<<<g, dim3(32, 1, 1), 0, stream>>>(in, wpk, out, Cin, Cout, Di, Di, Di);
  };
  auto stats = [&](const float* x, float* st, int C, int M) {
    bn_stats_kernel<<<dim3(C, 1, 1), dim3(256, 1, 1), 0, stream>>>(x, st, C, M);
  };
  auto bnap = [&](const float* x, float* st, const float* g, const float* b,
                  const unsigned short* idn, unsigned short* obf, float* of, int C, int M,
                  int Cstr, int coff, int relu) {
    long total = (long)C * M;
    bn_apply_kernel<<<dim3((unsigned)((total + 255) / 256), 1, 1), dim3(256, 1, 1), 0,
                      stream>>>(x, st, g, b, idn, obf, of, C, M, Cstr, coff, relu);
  };
  auto conv_bnrelu = [&](const unsigned short* in, int wi, int CinReal, int Cpad, int Cout,
                         int D, int k, int s, int p, unsigned short* obf) {
    conv(in, wi, scrA, CinReal, Cpad, Cout, D, k, s, p);
    int Do = (D + 2 * p - k) / s + 1;
    int Mo = Do * Do * Do;
    stats(scrA, st1, Cout, Mo);
    bnap(scrA, st1, Pf(wi + 1), Pf(wi + 2), nullptr, obf, nullptr, Cout, Mo, Cout, 0, 1);
  };
  auto run_block = [&](int bb, bool down, int cin, int planes, const unsigned short* xin,
                       int D, unsigned short* xout, float* xout_f) {
    int M = D * D * D;
    conv(xin, bb + 0, scrA, cin, cin, planes, D, 3, 1, 1);
    stats(scrA, st1, planes, M);
    bnap(scrA, st1, Pf(bb + 1), Pf(bb + 2), nullptr, bfT, nullptr, planes, M, planes, 0, 1);
    conv(bfT, bb + 3, scrA, planes, planes, planes, D, 3, 1, 1);
    stats(scrA, st2, planes, M);
    const unsigned short* idn = xin;
    if (down) {
      conv(xin, bb + 6, scrB, cin, cin, planes, D, 1, 1, 0);
      stats(scrB, st3, planes, M);
      bnap(scrB, st3, Pf(bb + 7), Pf(bb + 8), nullptr, bfI, nullptr, planes, M, planes, 0, 0);
      idn = bfI;
    }
    bnap(scrA, st2, Pf(bb + 4), Pf(bb + 5), idn, xout, xout_f, planes, M, planes, 0, 1);
  };
  auto run_stage = [&](int si, const unsigned short* xin, int D, unsigned short* xout,
                       float* xout_f) {
    int base = SBASE[si];
    bool down = SDOWN[si];
    run_block(base, down, SCIN[si], SPL[si], xin, D, bfU, nullptr);
    run_block(base + (down ? 9 : 6), false, SPL[si], SPL[si], bfU, D, xout, xout_f);
  };
  // deconv -> bn+relu into channels [0,Cout) of cat, then interleave skip
  auto up_cat = [&](const unsigned short* in, int tri, int Cin, int Cout, int Di,
                    const unsigned short* skip, int Cskip, unsigned short* cat) {
    deconv(in, tri, scrA, Cin, Cout, Di);
    int Mo = 8 * Di * Di * Di;
    int Ctot = Cout + Cskip;
    stats(scrA, st1, Cout, Mo);
    bnap(scrA, st1, Pf(tri + 1), Pf(tri + 2), nullptr, cat, nullptr, Cout, Mo, Ctot, 0, 1);
    long total = (long)Mo * Cskip;
    concat_copy_kernel<<<dim3((unsigned)((total + 255) / 256), 1, 1), dim3(256, 1, 1), 0,
                         stream>>>(skip, cat, Cskip, Ctot, Cout, Mo);
  };

  // =========================== forward pass ===============================
  cvt_x_kernel<<<dim3((unsigned)(((long)M48 * 32 + 255) / 256), 1, 1), dim3(256, 1, 1), 0,
                 stream>>>(Pf(0), xb, M48);

  // stem (k=5 pad=2, Cin 4 padded to 32) + bn + relu -> p1 skip 32@48^3
  conv_bnrelu(xb, I_C0, 4, 32, 32, 48, 5, 1, 2, p1);

  // encoder
  conv_bnrelu(p1, I_CV[1], 32, 32, 32, 48, 2, 2, 0, bufX);   // 32@24^3
  run_stage(0, bufX, 24, b1, nullptr);                       // out_b1 32@24^3
  conv_bnrelu(b1, I_CV[2], 32, 32, 32, 24, 2, 2, 0, bufX);   // 32@12^3
  run_stage(1, bufX, 12, b2, nullptr);                       // out_b2 64@12^3
  conv_bnrelu(b2, I_CV[3], 64, 64, 64, 12, 2, 2, 0, bufX);   // 64@6^3
  run_stage(2, bufX, 6, b3, nullptr);                        // out_b3 128@6^3
  conv_bnrelu(b3, I_CV[4], 128, 128, 128, 6, 2, 2, 0, bufX); // 128@3^3
  run_stage(3, bufX, 3, bufY, nullptr);                      // 256@3^3

  // decoder
  up_cat(bufY, I_TR[4], 256, 256, 3, b3, 128, bufX);   // cat 384@6^3
  run_stage(4, bufX, 6, bufY, nullptr);                // 256@6^3
  up_cat(bufY, I_TR[5], 256, 128, 6, b2, 64, bufX);    // cat 192@12^3
  run_stage(5, bufX, 12, bufY, nullptr);               // 128@12^3
  up_cat(bufY, I_TR[6], 128, 96, 12, b1, 32, bufX);    // cat 128@24^3
  run_stage(6, bufX, 24, bufY, nullptr);               // 96@24^3
  up_cat(bufY, I_TR[7], 96, 96, 24, p1, 32, bufX);     // cat 128@48^3

  // stage 7: features (fp32, channel-major) straight into d_out slot 2
  float* feat = (float*)d_out + (size_t)20 * M48;
  run_stage(7, bufX, 48, bufY, feat);                  // 96@48^3

  // final 1x1x1 conv -> logits (slot 1)
  final_conv_kernel<<<dim3((unsigned)(((long)20 * M48 + 255) / 256), 1, 1),
                      dim3(256, 1, 1), 0, stream>>>(feat, Pf(I_FW), Pf(I_FB),
                                                    (float*)d_out, M48);
}
